// MNELayer_84138409329109
// MI455X (gfx1250) — matrix operand order repaired and checked by reading, not compiled, not run on hardware
//
#include <hip/hip_runtime.h>

typedef __attribute__((ext_vector_type(16))) __bf16 v16bf;
typedef __attribute__((ext_vector_type(8)))  __bf16 v8bf;
typedef __attribute__((ext_vector_type(8)))  float  v8f;

#define F_DECAY 0.9f
#define F_SLOPE 0.01f
#define F_HEBB  0.01f
#define F_WD    1e-3f
#define F_EPS   1e-5f

static __device__ __forceinline__ v16bf make_frag(v8bf lo, v8bf hi) {
  union { v16bf v; v8bf h[2]; } u;
  u.h[0] = lo; u.h[1] = hi;
  return u.v;
}

// A fragment (16x32 bf16, wave32 layout): lane m = L&15, half = L>>4,
// elements 0..7  = A[m][k0 + 8*half + 0..7]
// elements 8..15 = A[m][k0 + 16 + 8*half + 0..7]
static __device__ __forceinline__ v16bf load_a(const __bf16* A, int lda,
                                               int m0, int k0, int lane) {
  const int m = m0 + (lane & 15);
  const int half = lane >> 4;
  const __bf16* p = A + (size_t)m * lda + k0 + 8 * half;
  return make_frag(*(const v8bf*)p, *(const v8bf*)(p + 16));
}

// B fragment (32x16 bf16): lane n = L&15, elements 0..7 = B[k0 + 8*half + e][n], 8..15 = B[k0 + 16 + 8*half + (e-8)][n].
// We feed B transposed row-major (BT[n][k]) so this is contiguous.
static __device__ __forceinline__ v16bf load_b(const __bf16* BT, int ldb,
                                               int n0, int k0, int lane) {
  const int n = n0 + (lane & 15);
  const int half = lane >> 4;
  const __bf16* p = BT + (size_t)n * ldb + k0 + 8 * half;
  return make_frag(*(const v8bf*)p, *(const v8bf*)(p + 16));
}

// ---------------- pre-norm + bf16 casts (one block per row, N = 4*256) -----
__global__ __launch_bounds__(256) void k_prenorm(const float* __restrict__ x,
                                                 const float* __restrict__ g,
                                                 const float* __restrict__ b,
                                                 __bf16* __restrict__ xnb,
                                                 __bf16* __restrict__ xb,
                                                 int N) {
  const int row = blockIdx.x, tid = threadIdx.x;
  __shared__ float s1[256], s2[256];
  const float4 v = ((const float4*)(x + (size_t)row * N))[tid];
  s1[tid] = v.x + v.y + v.z + v.w;
  s2[tid] = v.x * v.x + v.y * v.y + v.z * v.z + v.w * v.w;
  __syncthreads();
  for (int off = 128; off > 0; off >>= 1) {
    if (tid < off) { s1[tid] += s1[tid + off]; s2[tid] += s2[tid + off]; }
    __syncthreads();
  }
  const float mu = s1[0] / (float)N;
  const float var = s2[0] / (float)N - mu * mu;
  const float rstd = rsqrtf(var + F_EPS);
  const float xv[4] = {v.x, v.y, v.z, v.w};
  #pragma unroll
  for (int c = 0; c < 4; ++c) {
    const int j = tid * 4 + c;
    const float xn = (xv[c] - mu) * rstd * g[j] + b[j];
    xnb[(size_t)row * N + j] = (__bf16)xn;
    xb [(size_t)row * N + j] = (__bf16)xv[c];
  }
}

// ------------- weight prep: synWT[n][k] = syn[k][n]; gwb = cast(gate_W) ----
__global__ __launch_bounds__(256) void k_prepw(const float* __restrict__ syn,
                                               const float* __restrict__ gw,
                                               __bf16* __restrict__ synWT,
                                               __bf16* __restrict__ gwb,
                                               int N) {
  const int idx = blockIdx.x * 256 + threadIdx.x;   // idx = k*N + n
  const int k = idx / N, n = idx - k * N;
  synWT[(size_t)n * N + k] = (__bf16)syn[idx];      // coalesced read
  gwb[idx] = (__bf16)gw[idx];                       // gate_W[n][k] already == B^T
}

// --------- GEMM1: pre = xnorm @ syn; fused potential/act/actT epilogue -----
// block = 8 waves (4M x 2N), wave tile 64x64 (4x4 WMMA tiles), block 256x128
__global__ __launch_bounds__(256) void k_gemm_pre(const __bf16* __restrict__ A,
                                                  const __bf16* __restrict__ BT,
                                                  const float* __restrict__ prevp,
                                                  const float* __restrict__ contrib,
                                                  float* __restrict__ potential,
                                                  __bf16* __restrict__ actT,
                                                  __bf16* __restrict__ actcT,
                                                  int Bdim, int N) {
  const int lane = threadIdx.x & 31, wid = threadIdx.x >> 5;
  const int m_base = blockIdx.x * 256 + (wid >> 1) * 64;
  const int n_base = blockIdx.y * 128 + (wid & 1) * 64;
  const int K = N;
  v8f acc[4][4] = {};
  for (int k0 = 0; k0 < K; k0 += 32) {
    v16bf a[4], bb[4];
    #pragma unroll
    for (int i = 0; i < 4; ++i) a[i]  = load_a(A, K, m_base + 16 * i, k0, lane);
    #pragma unroll
    for (int j = 0; j < 4; ++j) bb[j] = load_b(BT, K, n_base + 16 * j, k0, lane);
    #pragma unroll
    for (int i = 0; i < 4; ++i)
      #pragma unroll
      for (int j = 0; j < 4; ++j)
        acc[i][j] = __builtin_amdgcn_wmma_f32_16x16x32_bf16(
            false, a[i], false, bb[j], (short)0, acc[i][j], false, false);
  }
  const int lane15 = lane & 15, half = lane >> 4;
  #pragma unroll
  for (int i = 0; i < 4; ++i) {
    const int mrow0 = m_base + 16 * i + 8 * half;
    #pragma unroll
    for (int j = 0; j < 4; ++j) {
      const int n = n_base + 16 * j + lane15;
      v8bf av, acv;
      #pragma unroll
      for (int r = 0; r < 8; ++r) {
        const int m = mrow0 + r;
        const float pot = F_DECAY * prevp[(size_t)m * N + n] + acc[i][j][r];
        potential[(size_t)m * N + n] = pot;
        const float a = pot >= 0.f ? pot : F_SLOPE * pot;
        av[r]  = (__bf16)a;
        acv[r] = (__bf16)(a * contrib[m]);
      }
      *(v8bf*)(actT  + (size_t)n * Bdim + mrow0) = av;   // transposed, contiguous
      *(v8bf*)(actcT + (size_t)n * Bdim + mrow0) = acv;
    }
  }
}

// ------ GEMM3: z = inputs @ gate_W^T; fused sigmoid + gated residual -------
__global__ __launch_bounds__(256) void k_gemm_gate(const __bf16* __restrict__ A,
                                                   const __bf16* __restrict__ BT,
                                                   const float* __restrict__ gate_b,
                                                   const float* __restrict__ potential,
                                                   const float* __restrict__ xin,
                                                   float* __restrict__ resid,
                                                   int Bdim, int N) {
  const int lane = threadIdx.x & 31, wid = threadIdx.x >> 5;
  const int m_base = blockIdx.x * 256 + (wid >> 1) * 64;
  const int n_base = blockIdx.y * 128 + (wid & 1) * 64;
  const int K = N;
  v8f acc[4][4] = {};
  for (int k0 = 0; k0 < K; k0 += 32) {
    v16bf a[4], bb[4];
    #pragma unroll
    for (int i = 0; i < 4; ++i) a[i]  = load_a(A, K, m_base + 16 * i, k0, lane);
    #pragma unroll
    for (int j = 0; j < 4; ++j) bb[j] = load_b(BT, K, n_base + 16 * j, k0, lane);
    #pragma unroll
    for (int i = 0; i < 4; ++i)
      #pragma unroll
      for (int j = 0; j < 4; ++j)
        acc[i][j] = __builtin_amdgcn_wmma_f32_16x16x32_bf16(
            false, a[i], false, bb[j], (short)0, acc[i][j], false, false);
  }
  const int lane15 = lane & 15, half = lane >> 4;
  #pragma unroll
  for (int i = 0; i < 4; ++i) {
    const int mrow0 = m_base + 16 * i + 8 * half;
    #pragma unroll
    for (int j = 0; j < 4; ++j) {
      const int n = n_base + 16 * j + lane15;
      const float gb = gate_b[n];
      #pragma unroll
      for (int r = 0; r < 8; ++r) {
        const int m = mrow0 + r;
        const float z = acc[i][j][r] + gb;
        const float g = 1.f / (1.f + __expf(-z));
        const float pot = potential[(size_t)m * N + n];
        const float a = pot >= 0.f ? pot : F_SLOPE * pot;
        const float x = xin[(size_t)m * N + n];
        resid[(size_t)m * N + n] = g * a + (1.f - g) * x;
      }
    }
  }
}

// ------------- post-norm, in place on the residual in d_out[0] -------------
__global__ __launch_bounds__(256) void k_postnorm(float* __restrict__ y,
                                                  const float* __restrict__ g,
                                                  const float* __restrict__ b,
                                                  int N) {
  const int row = blockIdx.x, tid = threadIdx.x;
  __shared__ float s1[256], s2[256];
  const float4 v = ((const float4*)(y + (size_t)row * N))[tid];
  s1[tid] = v.x + v.y + v.z + v.w;
  s2[tid] = v.x * v.x + v.y * v.y + v.z * v.z + v.w * v.w;
  __syncthreads();
  for (int off = 128; off > 0; off >>= 1) {
    if (tid < off) { s1[tid] += s1[tid + off]; s2[tid] += s2[tid + off]; }
    __syncthreads();
  }
  const float mu = s1[0] / (float)N;
  const float var = s2[0] / (float)N - mu * mu;
  const float rstd = rsqrtf(var + F_EPS);
  const float xv[4] = {v.x, v.y, v.z, v.w};
  #pragma unroll
  for (int c = 0; c < 4; ++c) {
    const int j = tid * 4 + c;
    y[(size_t)row * N + j] = (xv[c] - mu) * rstd * g[j] + b[j];
  }
}

// -------- GEMM2: hebb = actc^T @ act / B; fused decay + clip ---------------
// A = actcT [N][B] row-major (rows of actc^T are contiguous), BT = actT.
// block = 8 waves (4M x 2N), wave tile 32x32 (2x2), block 128x64.
__global__ __launch_bounds__(256) void k_gemm_hebb(const __bf16* __restrict__ A,
                                                   const __bf16* __restrict__ BT,
                                                   const float* __restrict__ syn,
                                                   float* __restrict__ out,
                                                   int Nn, int Bdim) {
  const int lane = threadIdx.x & 31, wid = threadIdx.x >> 5;
  const int m_base = blockIdx.x * 128 + (wid >> 1) * 32;
  const int n_base = blockIdx.y * 64 + (wid & 1) * 32;
  v8f acc[2][2] = {};
  for (int k0 = 0; k0 < Bdim; k0 += 32) {
    v16bf a[2], bb[2];
    #pragma unroll
    for (int i = 0; i < 2; ++i) a[i]  = load_a(A, Bdim, m_base + 16 * i, k0, lane);
    #pragma unroll
    for (int j = 0; j < 2; ++j) bb[j] = load_b(BT, Bdim, n_base + 16 * j, k0, lane);
    #pragma unroll
    for (int i = 0; i < 2; ++i)
      #pragma unroll
      for (int j = 0; j < 2; ++j)
        acc[i][j] = __builtin_amdgcn_wmma_f32_16x16x32_bf16(
            false, a[i], false, bb[j], (short)0, acc[i][j], false, false);
  }
  const int lane15 = lane & 15, half = lane >> 4;
  const float hs = F_HEBB / (float)Bdim;
  #pragma unroll
  for (int i = 0; i < 2; ++i) {
    const int mrow0 = m_base + 16 * i + 8 * half;
    #pragma unroll
    for (int j = 0; j < 2; ++j) {
      const int n = n_base + 16 * j + lane15;
      #pragma unroll
      for (int r = 0; r < 8; ++r) {
        const int m = mrow0 + r;
        float w = syn[(size_t)m * Nn + n] * (1.f - F_WD) + hs * acc[i][j][r];
        w = fminf(1.f, fmaxf(-1.f, w));
        out[(size_t)m * Nn + n] = w;
      }
    }
  }
}

extern "C" void kernel_launch(void* const* d_in, const int* in_sizes, int n_in,
                              void* d_out, int out_size, void* d_ws, size_t ws_size,
                              hipStream_t stream) {
  const float* inputs   = (const float*)d_in[0];
  const float* contrib  = (const float*)d_in[1];
  const float* prevp    = (const float*)d_in[2];
  const float* syn      = (const float*)d_in[3];
  const float* gateW    = (const float*)d_in[4];
  const float* gate_b   = (const float*)d_in[5];
  const float* pre_g    = (const float*)d_in[6];
  const float* pre_b    = (const float*)d_in[7];
  const float* ln_g     = (const float*)d_in[8];
  const float* ln_b     = (const float*)d_in[9];

  const int B = in_sizes[1];            // 16384
  const int N = in_sizes[0] / B;        // 1024
  const size_t BN = (size_t)B * N, NN = (size_t)N * N;

  float* out0      = (float*)d_out;          // output  [B,N]
  float* potential = out0 + BN;              // potential [B,N]
  float* out_syn   = potential + BN;         // new_syn [N,N]

  char* ws = (char*)d_ws;
  __bf16* xnormb = (__bf16*)ws;              ws += BN * 2;
  __bf16* inb    = (__bf16*)ws;              ws += BN * 2;
  __bf16* synWT  = (__bf16*)ws;              ws += NN * 2;
  __bf16* gwb    = (__bf16*)ws;              ws += NN * 2;
  __bf16* actT   = (__bf16*)ws;              ws += BN * 2;
  __bf16* actcT  = (__bf16*)ws;              ws += BN * 2;

  k_prenorm<<<B, 256, 0, stream>>>(inputs, pre_g, pre_b, xnormb, inb, N);
  k_prepw<<<(unsigned)(NN / 256), 256, 0, stream>>>(syn, gateW, synWT, gwb, N);

  dim3 g1(B / 256, N / 128);
  k_gemm_pre<<<g1, 256, 0, stream>>>(xnormb, synWT, prevp, contrib,
                                     potential, actT, actcT, B, N);
  k_gemm_gate<<<g1, 256, 0, stream>>>(inb, gwb, gate_b, potential, inputs,
                                      out0, B, N);
  k_postnorm<<<B, 256, 0, stream>>>(out0, ln_g, ln_b, N);

  dim3 g2(N / 128, N / 64);
  k_gemm_hebb<<<g2, 256, 0, stream>>>(actcT, actT, syn, out_syn, N, B);
}
